// Simple_MP_Layer_36189394436111
// MI455X (gfx1250) — matrix operand order repaired and checked by reading, not compiled, not run on hardware
//
#include <hip/hip_runtime.h>

typedef __attribute__((ext_vector_type(16))) __bf16 v16bf;
typedef __attribute__((ext_vector_type(8)))  __bf16 v8bf;
typedef __attribute__((ext_vector_type(8)))  float  v8f;
typedef __attribute__((ext_vector_type(4)))  float  v4f;
typedef __attribute__((ext_vector_type(4)))  int    v4i;

static constexpr int HD      = 128;      // feature width
static constexpr int NNODES  = 50000;
static constexpr int NEDGES  = 800000;

// ---- d_ws layout ----
// [floats]  agg : NNODES*HD          (scatter-sum accumulator)
// [floats]  cnt : NNODES             (scatter counts)
// [bf16]    packed weight fragments  (fragment-major, see pack_weights)
static constexpr size_t AGG_ELEMS   = (size_t)NNODES * HD;          // 6,400,000
static constexpr size_t ZERO_ELEMS  = AGG_ELEMS + NNODES;           // 6,450,000
static constexpr size_t WFRAG_OFF_B = ZERO_ELEMS * sizeof(float);   // 25,800,000 (32B aligned)

// fragment-area offsets in bf16 elements: frag(ks,nt) = 512 bf16 (32 lanes x 16)
static constexpr int WE0_F = 0;       // w_e0: 12 K-steps (K=384)  -> 49152 elems (96 KB)
static constexpr int WE1_F = 49152;   // w_e1:  4 K-steps (K=128)
static constexpr int WN0_F = 65536;   // w_n0:  8 K-steps (K=256)  -> 32768 elems (64 KB)
static constexpr int WN1_F = 98304;   // w_n1:  4 K-steps (K=128)

// ---------------------------------------------------------------------------
// CDNA5 async global->LDS copy (GLOBAL_LOAD_ASYNC_TO_LDS_B128, ASYNCcnt),
// with a portable fallback (global load + ds_store) if the builtin is absent.
// Builtin params are typed: (v4i __device__*, v4i __shared__*, Ii offset, Ii cpol)
// ---------------------------------------------------------------------------
#if defined(__has_builtin)
#  if __has_builtin(__builtin_amdgcn_global_load_async_to_lds_b128)
#    define HAVE_ASYNC_LDS 1
#  endif
#endif
#ifndef HAVE_ASYNC_LDS
#  define HAVE_ASYNC_LDS 0
#endif

__device__ __forceinline__ void copy16_to_lds(const __bf16* g, __bf16* l) {
#if HAVE_ASYNC_LDS
  typedef __attribute__((address_space(1))) v4i*  gp_t;
  typedef __attribute__((address_space(3))) v4i*  lp_t;
  typedef __attribute__((address_space(1))) void* gv_t;
  typedef __attribute__((address_space(3))) void* lv_t;
  __builtin_amdgcn_global_load_async_to_lds_b128(
      (gp_t)(gv_t)(void*)g,
      (lp_t)(lv_t)(void*)l, 0, 0);
#else
  *(v8bf*)l = *(const v8bf*)g;
#endif
}

__device__ __forceinline__ void wait_lds_copies() {
#if HAVE_ASYNC_LDS
  asm volatile("s_wait_asynccnt 0" ::: "memory");   // SOPP opcode 74 (gfx1250)
#endif
}

// ---------------------------------------------------------------------------
// Zero the scatter accumulators (must run every call: ws is not re-poisoned)
// ---------------------------------------------------------------------------
__global__ void zero_f32(float* __restrict__ p, int n) {
  int i = blockIdx.x * blockDim.x + threadIdx.x;
  if (i < n) p[i] = 0.0f;
}

// ---------------------------------------------------------------------------
// Repack row-major fp32 W[K x 128] into bf16 WMMA B fragments.
// B order used here (contiguous k, same as load_a_frag): lane l, element i ->
//   n = nt*16 + (l & 15),  k = ks*32 + (l >> 4)*16 + i
// Stored fragment-major so each lane's fragment is 32 contiguous bytes.
// ---------------------------------------------------------------------------
__global__ void pack_weights(const float* __restrict__ W, __bf16* __restrict__ dst,
                             int ksteps) {
  int idx = blockIdx.x * blockDim.x + threadIdx.x;
  int total = ksteps * 8 * 512;
  if (idx >= total) return;
  int f    = idx >> 9;          // fragment id = ks*8 + nt
  int rem  = idx & 511;
  int lane = rem >> 4;
  int i    = rem & 15;
  int ks   = f >> 3;
  int nt   = f & 7;
  int k = ks * 32 + (lane >> 4) * 16 + i;
  int n = nt * 16 + (lane & 15);
  dst[idx] = (__bf16)W[k * HD + n];
}

// ---------------------------------------------------------------------------
// A-fragment gather from a bf16 LDS tile row, column base cb.
// A order used here (contiguous k, same as pack_weights): lane l (m = l&15, khalf = l>>4):
//   elems 0..7  -> K = khalf*16 + 0..7
//   elems 8..15 -> K = khalf*16 + 8 + 0..7
// ---------------------------------------------------------------------------
__device__ __forceinline__ v16bf load_a_frag(const __bf16* row, int cb, int khalf) {
  union { v16bf v; v8bf h[2]; } A;
  A.h[0] = *(const v8bf*)(row + cb + khalf * 16);
  A.h[1] = *(const v8bf*)(row + cb + khalf * 16 + 8);
  return A.v;
}

__device__ __forceinline__ float elu(float v) {
  return v > 0.0f ? v : (__expf(v) - 1.0f);
}

// ---------------------------------------------------------------------------
// Edge kernel: 4 waves/block, 16 edges per wave (64/block, 12500 blocks).
//   w_e0 (96 KB bf16 fragments) staged block-wide in LDS via async DMA.
//   feat = [x_own | x_nei | diff]  (16 x 384 bf16, staged per-wave in LDS)
//   h    = ELU(feat @ w_e0)        (12 K-steps of WMMA, B from LDS)
//   e    = diff + h @ w_e1         ( 4 K-steps of WMMA, B from L2)
//   atomicAdd into agg[dst], cnt[dst]
// ---------------------------------------------------------------------------
__global__ __launch_bounds__(128)
void edge_kernel(const float* __restrict__ x, const int* __restrict__ ei,
                 const __bf16* __restrict__ wf,
                 float* __restrict__ agg, float* __restrict__ cnt) {
  __shared__ __bf16 s_we0[12 * 8 * 512];       // 96 KB: w_e0 fragments
  __shared__ __bf16 s_own [4][16][HD];         // 16 KB
  __shared__ __bf16 s_nei [4][16][HD];         // 16 KB
  __shared__ __bf16 s_diff[4][16][HD];         // 16 KB
  __shared__ int    s_src [4][16];
  __shared__ int    s_dst [4][16];

  const int t     = threadIdx.x;
  const int lane  = t & 31;
  const int w     = t >> 5;
  const int mrow  = lane & 15;
  const int khalf = lane >> 4;
  const int base  = blockIdx.x * 64 + w * 16;

  // ---- kick off block-wide async DMA of w_e0 fragments into LDS ----
  // 49152 bf16 = 6144 x 16B chunks; 128 threads x 48 chunks
#pragma unroll 4
  for (int c = 0; c < 48; ++c) {
    int idx = (c * 128 + t) * 8;               // 8 bf16 per 16B chunk
    copy16_to_lds(wf + WE0_F + idx, &s_we0[idx]);
  }
  __builtin_prefetch(wf + WE1_F + (size_t)t * 128, 0, 0);  // warm L2 for GEMM2 weights

  // ---- stage per-row edge indices (lanes 0..15), count atomics ----
  if (lane < 16) {
    int e = base + lane;
    int s = ei[e];
    int d = ei[NEDGES + e];
    s_src[w][lane] = s;
    s_dst[w][lane] = d;
    atomicAdd(&cnt[d], 1.0f);
  }
  // wave-internal LDS ordering: DS ops from one wave complete in order.

  // ---- stage x_own / x_nei / diff rows as bf16 (each lane: 4 cols/row) ----
  for (int r = 0; r < 16; ++r) {
    int s = s_src[w][r];
    int d = s_dst[w][r];
    v4f a = *(const v4f*)(x + (size_t)s * HD + lane * 4);
    v4f b = *(const v4f*)(x + (size_t)d * HD + lane * 4);
#pragma unroll
    for (int q = 0; q < 4; ++q) {
      s_own [w][r][lane * 4 + q] = (__bf16)a[q];
      s_nei [w][r][lane * 4 + q] = (__bf16)b[q];
      s_diff[w][r][lane * 4 + q] = (__bf16)(a[q] - b[q]);
    }
  }

  // ---- make DMA'd weights visible block-wide ----
  wait_lds_copies();
  __syncthreads();

  // ---- GEMM1: [16 x 384] @ [384 x 128] -> acc (f32), B from LDS ----
  v8f acc[8];
#pragma unroll
  for (int nt = 0; nt < 8; ++nt) acc[nt] = v8f{0,0,0,0,0,0,0,0};

#pragma unroll
  for (int ks = 0; ks < 12; ++ks) {
    const __bf16* row = (ks < 4) ? &s_own [w][mrow][0]
                      : (ks < 8) ? &s_nei [w][mrow][0]
                                 : &s_diff[w][mrow][0];
    v16bf A = load_a_frag(row, (ks & 3) * 32, khalf);
#pragma unroll
    for (int nt = 0; nt < 8; ++nt) {
      v16bf B = *(const v16bf*)&s_we0[(ks * 8 + nt) * 512 + lane * 16];
      acc[nt] = __builtin_amdgcn_wmma_f32_16x16x32_bf16(false, A, false, B,
                                                        (short)0, acc[nt], false, false);
    }
  }

  // ---- ELU, re-layout C->A through LDS (reuse s_own; wave-local) ----
  // C/D layout: lane l holds (m = khalf*8 + j, n = nt*16 + mrow)
#pragma unroll
  for (int nt = 0; nt < 8; ++nt)
#pragma unroll
    for (int j = 0; j < 8; ++j)
      s_own[w][khalf * 8 + j][nt * 16 + mrow] = (__bf16)elu(acc[nt][j]);

  // ---- GEMM2: [16 x 128] @ [128 x 128] -> acc2, B from L2 ----
  v8f acc2[8];
#pragma unroll
  for (int nt = 0; nt < 8; ++nt) acc2[nt] = v8f{0,0,0,0,0,0,0,0};

#pragma unroll
  for (int ks = 0; ks < 4; ++ks) {
    v16bf A = load_a_frag(&s_own[w][mrow][0], ks * 32, khalf);
#pragma unroll
    for (int nt = 0; nt < 8; ++nt) {
      v16bf B = *(const v16bf*)(wf + WE1_F + (size_t)(ks * 8 + nt) * 512 + lane * 16);
      acc2[nt] = __builtin_amdgcn_wmma_f32_16x16x32_bf16(false, A, false, B,
                                                         (short)0, acc2[nt], false, false);
    }
  }

  // ---- residual + scatter-add into L2-resident accumulator ----
#pragma unroll
  for (int nt = 0; nt < 8; ++nt)
#pragma unroll
    for (int j = 0; j < 8; ++j) {
      int m = khalf * 8 + j;
      int n = nt * 16 + mrow;
      float v = (float)s_diff[w][m][n] + acc2[nt][j];
      int d = s_dst[w][m];
      atomicAdd(&agg[(size_t)d * HD + n], v);
    }
}

// ---------------------------------------------------------------------------
// Node kernel: 4 waves/block, 16 nodes per wave.
//   w_n0 (64 KB bf16 fragments) staged block-wide in LDS via async DMA.
//   feat = [x | agg/max(cnt,1)]  (16 x 256)
//   out  = x + ELU(feat @ w_n0) @ w_n1
// ---------------------------------------------------------------------------
__global__ __launch_bounds__(128)
void node_kernel(const float* __restrict__ x, const __bf16* __restrict__ wf,
                 const float* __restrict__ agg, const float* __restrict__ cnt,
                 float* __restrict__ out) {
  __shared__ __bf16 s_wn0[8 * 8 * 512];        // 64 KB: w_n0 fragments
  __shared__ __bf16 s_feat[4][16][2 * HD];     // 32 KB
  __shared__ float  s_xf  [4][16][HD];         // 32 KB
  __shared__ __bf16 s_h   [4][16][HD];         // 16 KB

  const int t     = threadIdx.x;
  const int lane  = t & 31;
  const int w     = t >> 5;
  const int mrow  = lane & 15;
  const int khalf = lane >> 4;
  const int base  = blockIdx.x * 64 + w * 16;

  // ---- async DMA of w_n0 fragments: 32768 bf16 = 4096 x 16B chunks ----
#pragma unroll 4
  for (int c = 0; c < 32; ++c) {
    int idx = (c * 128 + t) * 8;
    copy16_to_lds(wf + WN0_F + idx, &s_wn0[idx]);
  }
  __builtin_prefetch(wf + WN1_F + (size_t)t * 128, 0, 0);

  // ---- stage [x | mean-agg] as bf16, keep x in f32 for the residual ----
  for (int r = 0; r < 16; ++r) {
    int node = base + r;
    int nn = node < NNODES ? node : 0;          // clamp (tail rows recomputed, not stored)
    v4f a = *(const v4f*)(x   + (size_t)nn * HD + lane * 4);
    v4f g = *(const v4f*)(agg + (size_t)nn * HD + lane * 4);
    float inv = 1.0f / fmaxf(cnt[nn], 1.0f);
#pragma unroll
    for (int q = 0; q < 4; ++q) {
      s_feat[w][r][lane * 4 + q]      = (__bf16)a[q];
      s_feat[w][r][HD + lane * 4 + q] = (__bf16)(g[q] * inv);
      s_xf  [w][r][lane * 4 + q]      = a[q];
    }
  }

  wait_lds_copies();
  __syncthreads();

  // ---- GEMM1: [16 x 256] @ [256 x 128], B from LDS ----
  v8f acc[8];
#pragma unroll
  for (int nt = 0; nt < 8; ++nt) acc[nt] = v8f{0,0,0,0,0,0,0,0};

#pragma unroll
  for (int ks = 0; ks < 8; ++ks) {
    v16bf A = load_a_frag(&s_feat[w][mrow][0], ks * 32, khalf);
#pragma unroll
    for (int nt = 0; nt < 8; ++nt) {
      v16bf B = *(const v16bf*)&s_wn0[(ks * 8 + nt) * 512 + lane * 16];
      acc[nt] = __builtin_amdgcn_wmma_f32_16x16x32_bf16(false, A, false, B,
                                                        (short)0, acc[nt], false, false);
    }
  }

  // ---- ELU -> LDS (C layout -> A layout) ----
#pragma unroll
  for (int nt = 0; nt < 8; ++nt)
#pragma unroll
    for (int j = 0; j < 8; ++j)
      s_h[w][khalf * 8 + j][nt * 16 + mrow] = (__bf16)elu(acc[nt][j]);

  // ---- GEMM2: [16 x 128] @ [128 x 128], B from L2 ----
  v8f acc2[8];
#pragma unroll
  for (int nt = 0; nt < 8; ++nt) acc2[nt] = v8f{0,0,0,0,0,0,0,0};

#pragma unroll
  for (int ks = 0; ks < 4; ++ks) {
    v16bf A = load_a_frag(&s_h[w][mrow][0], ks * 32, khalf);
#pragma unroll
    for (int nt = 0; nt < 8; ++nt) {
      v16bf B = *(const v16bf*)(wf + WN1_F + (size_t)(ks * 8 + nt) * 512 + lane * 16);
      acc2[nt] = __builtin_amdgcn_wmma_f32_16x16x32_bf16(false, A, false, B,
                                                         (short)0, acc2[nt], false, false);
    }
  }

  // ---- residual + store (mask tail rows; WMMAs above ran with full EXEC) ----
#pragma unroll
  for (int nt = 0; nt < 8; ++nt)
#pragma unroll
    for (int j = 0; j < 8; ++j) {
      int m = khalf * 8 + j;
      int row = base + m;
      if (row < NNODES) {
        int n = nt * 16 + mrow;
        out[(size_t)row * HD + n] = s_xf[w][m][n] + acc2[nt][j];
      }
    }
}

// ---------------------------------------------------------------------------
extern "C" void kernel_launch(void* const* d_in, const int* in_sizes, int n_in,
                              void* d_out, int out_size, void* d_ws, size_t ws_size,
                              hipStream_t stream) {
  (void)in_sizes; (void)n_in; (void)out_size; (void)ws_size;
  const float* x    = (const float*)d_in[0];
  const int*   ei   = (const int*)  d_in[1];
  const float* we0  = (const float*)d_in[2];
  const float* we1  = (const float*)d_in[3];
  const float* wn0  = (const float*)d_in[4];
  const float* wn1  = (const float*)d_in[5];
  float*       out  = (float*)d_out;

  float*  agg = (float*)d_ws;
  float*  cnt = agg + AGG_ELEMS;
  __bf16* wf  = (__bf16*)((char*)d_ws + WFRAG_OFF_B);

  // 1) zero scatter accumulators (every call; ws is dirty between replays)
  {
    int n = (int)ZERO_ELEMS;
    zero_f32<<<(n + 255) / 256, 256, 0, stream>>>(agg, n);
  }
  // 2) repack weights to bf16 WMMA B fragments
  pack_weights<<<(12 * 8 * 512 + 255) / 256, 256, 0, stream>>>(we0, wf + WE0_F, 12);
  pack_weights<<<( 4 * 8 * 512 + 255) / 256, 256, 0, stream>>>(we1, wf + WE1_F,  4);
  pack_weights<<<( 8 * 8 * 512 + 255) / 256, 256, 0, stream>>>(wn0, wf + WN0_F,  8);
  pack_weights<<<( 4 * 8 * 512 + 255) / 256, 256, 0, stream>>>(wn1, wf + WN1_F,  4);
  // 3) fused edge MLP + scatter-add
  edge_kernel<<<NEDGES / 64, 128, 0, stream>>>(x, ei, wf, agg, cnt);
  // 4) fused node MLP + residual
  node_kernel<<<(NNODES + 63) / 64, 128, 0, stream>>>(x, wf, agg, cnt, out);
}
